// RNNModelOld_27771258536198
// MI455X (gfx1250) — compile-verified, matrix operand order checked by reading, not run on hardware
//
#include <hip/hip_runtime.h>

// ---- problem dims ----
#define B_  256
#define T_  1024
#define OB_ 128
#define AC_ 32
#define S_  512
#define H_  1024

// LDS row pitches (bf16 elements). pitch % 8 == 0 keeps rows 16B aligned.
#define XP 520
#define YP 136
#define ZP 1032

typedef __attribute__((ext_vector_type(16))) __bf16 v16bf;
typedef __attribute__((ext_vector_type(8)))  __bf16 v8bf;
typedef __attribute__((ext_vector_type(8)))  float  v8f;

static __device__ __forceinline__ v8f wmma_bf16(v16bf a, v16bf b, v8f c) {
  // D = A(16x32 bf16) x B(32x16 bf16) + C(16x16 f32)
  return __builtin_amdgcn_wmma_f32_16x16x32_bf16(
      /*neg_a=*/false, a, /*neg_b=*/false, b,
      /*c_mod=*/(short)0, c, /*reuse_a=*/false, /*reuse_b=*/false);
}

// gfx1250 hardware tanh (V_TANH_F32, single TRANS op) when available.
static __device__ __forceinline__ float fast_tanh(float x) {
#if __has_builtin(__builtin_amdgcn_tanhf)
  return __builtin_amdgcn_tanhf(x);
#else
  return tanhf(x);
#endif
}

// A-matrix fragment (16-bit, 16x32): lanes 0-15 hold row m, K = kb..kb+7 and
// kb+16..kb+23; lanes 16-31 same rows with kb offset +8. Two 16B LDS loads.
static __device__ __forceinline__ v16bf load_afrag(const __bf16* row, int kb) {
  v8bf lo = *(const v8bf*)(row + kb);
  v8bf hi = *(const v8bf*)(row + kb + 16);
  return __builtin_shufflevector(lo, hi, 0,1,2,3,4,5,6,7,8,9,10,11,12,13,14,15);
}

// B-matrix fragment: pre-packed in lane order (32B contiguous per lane).
static __device__ __forceinline__ v16bf load_wfrag(const __bf16* base, int tile, int lane) {
  return *((const v16bf*)(base + (size_t)tile * 512) + lane);
}

// K-tile accumulate loop; unroll_count(4) = 4 fragment pairs in flight
// (~64 VGPRs of staging + 8 accumulator regs, fine at the 256-VGPR budget
// of a 512-thread block) without triggering the full-unroll spill storm.
static __device__ __forceinline__ v8f gemm_acc(v8f acc, const __bf16* row, int hi8off,
                                               const __bf16* wbase, int nt, int KT,
                                               int lane) {
#pragma clang loop unroll_count(4)
  for (int kt = 0; kt < KT; ++kt) {
    v16bf a = load_afrag(row, kt * 32 + hi8off);
    v16bf b = load_wfrag(wbase, nt * KT + kt, lane);
    acc = wmma_bf16(a, b, acc);
  }
  return acc;
}

// Pre-pack a KxN f32 weight (used as B operand, row index = K) into bf16
// WMMA B-fragments: frag[nt*KT + kt][lane][e], where for lane l, element e:
//   n  = nt*16 + (l & 15)
//   kb = kt*32 + ((l>>4)&1)*8
//   k  = (e < 8) ? kb + e : kb + 8 + e
__global__ void pack_b_frags(const float* __restrict__ W, __bf16* __restrict__ dst,
                             int K, int N) {
  int tid = blockIdx.x * blockDim.x + threadIdx.x;
  int total = K * N;
  if (tid >= total) return;
  int KT   = K >> 5;
  int tile = tid >> 9;          // 512 elements per tile
  int lane = (tid >> 4) & 31;
  int e    = tid & 15;
  int nt   = tile / KT;
  int kt   = tile - nt * KT;
  int n    = nt * 16 + (lane & 15);
  int kb   = kt * 32 + ((lane >> 4) & 1) * 8;
  int k    = (e < 8) ? (kb + e) : (kb + 8 + e);
  dst[tid] = (__bf16)W[(size_t)k * N + n];
}

// One workgroup per 16 batch rows; 16 waves split GEMM N-dims.
// The ξ recurrence is per batch row, so no cross-WG sync is needed.
__global__ __launch_bounds__(512)
void rnn_scan_kernel(const float* __restrict__ obs,
                     const float* __restrict__ state0,
                     const float* __restrict__ log_stds,
                     const __bf16* __restrict__ ck2f, const __bf16* __restrict__ dk4f,
                     const __bf16* __restrict__ ck1f, const __bf16* __restrict__ dk1f,
                     const __bf16* __restrict__ dk2f, const __bf16* __restrict__ akf,
                     const __bf16* __restrict__ bk1f, const __bf16* __restrict__ bk2f,
                     float* __restrict__ out) {
  __shared__ __align__(16) __bf16 xib[16 * XP];   // ξ  [16 x 512] bf16
  __shared__ __align__(16) __bf16 yb [16 * YP];   // y  [16 x 128] bf16
  __shared__ __align__(16) __bf16 zb [16 * ZP];   // z  [16 x 1024] bf16

  const int tid  = threadIdx.x;
  const int wave = tid >> 5;        // 0..15
  const int lane = tid & 31;
  const int m0   = blockIdx.x * 16; // batch tile base
  const int mrow = lane & 15;
  const int hi8  = ((lane >> 4) & 1) * 8;

  // init ξ from input state (f32 -> bf16)
  for (int e = tid; e < 16 * S_; e += 512) {
    int m = e >> 9;
    int s = e & (S_ - 1);
    xib[m * XP + s] = (__bf16)state0[(size_t)(m0 + m) * S_ + s];
  }

  const float logv = log_stds[lane & (AC_ - 1)];

  const __bf16* xrow = xib + mrow * XP;
  const __bf16* yrow = yb  + mrow * YP;
  const __bf16* zrow = zb  + mrow * ZP;
  const size_t  out_fs_base = (size_t)B_ * T_ * (2 * AC_);

  for (int t = 0; t < T_; ++t) {
    __syncthreads();   // ξ stores from previous step visible; yb free to overwrite

    // stage y tile: 16x128 f32 -> bf16 in LDS (4 elements / thread)
    {
      int e0 = tid * 4;
      int m  = e0 >> 7;
      int ob = e0 & (OB_ - 1);
      const float* src = obs + ((size_t)(m0 + m) * T_ + t) * OB_ + ob;
      float4 f = *(const float4*)src;
      yb[m * YP + ob]     = (__bf16)f.x;
      yb[m * YP + ob + 1] = (__bf16)f.y;
      yb[m * YP + ob + 2] = (__bf16)f.z;
      yb[m * YP + ob + 3] = (__bf16)f.w;
    }
    __syncthreads();

    // ---- Phase A: v = ξ@CK2 + y@DK4 ; z = 2*tanh(v) - v ----
#pragma clang loop unroll(disable)
    for (int i = 0; i < 4; ++i) {
      int nt = wave + 16 * i;                          // H/16 = 64 tiles
      v8f acc = {};
      acc = gemm_acc(acc, xrow, hi8, ck2f, nt, S_ / 32, lane);
      acc = gemm_acc(acc, yrow, hi8, dk4f, nt, OB_ / 32, lane);
      int ncol = nt * 16 + (lane & 15);
#pragma unroll
      for (int r = 0; r < 8; ++r) {                    // C layout: m=r+hi8, n=lane&15
        int   m = r + hi8;
        float v = acc[r];
        zb[m * ZP + ncol] = (__bf16)(2.0f * fast_tanh(v) - v);
      }
    }
    __syncthreads();

    // ---- Phase B ----
    // u = ξ@CK1 + z@DK1 + y@DK2  (waves 0,1; AC/16 = 2 tiles), stored to out
    if (wave < 2) {
      int nt = wave;
      v8f acc = {};
      acc = gemm_acc(acc, xrow, hi8, ck1f, nt, S_ / 32, lane);
      acc = gemm_acc(acc, zrow, hi8, dk1f, nt, H_ / 32, lane);
      acc = gemm_acc(acc, yrow, hi8, dk2f, nt, OB_ / 32, lane);
      int c = nt * 16 + (lane & 15);
#pragma unroll
      for (int r = 0; r < 8; ++r) {
        int m = r + hi8;
        out[((size_t)(m0 + m) * T_ + t) * (2 * AC_) + c] = acc[r];
      }
    } else if (wave < 4) {
      // second half of outputs: broadcast log_stds
      int rbase = (wave - 2) * 8;
#pragma unroll
      for (int j = 0; j < 8; ++j) {
        int m = rbase + j;
        out[((size_t)(m0 + m) * T_ + t) * (2 * AC_) + AC_ + lane] = logv;
      }
    }

    // ξ_next = ξ@AK + z@BK1 + y@BK2  (S/16 = 32 tiles, two per wave);
    // hold in accumulators across the barrier, then overwrite ξ in place.
    v8f xacc0 = {}, xacc1 = {};
    {
      int nt0 = wave, nt1 = wave + 16;
      xacc0 = gemm_acc(xacc0, xrow, hi8, akf,  nt0, S_ / 32,  lane);
      xacc0 = gemm_acc(xacc0, zrow, hi8, bk1f, nt0, H_ / 32,  lane);
      xacc0 = gemm_acc(xacc0, yrow, hi8, bk2f, nt0, OB_ / 32, lane);
      xacc1 = gemm_acc(xacc1, xrow, hi8, akf,  nt1, S_ / 32,  lane);
      xacc1 = gemm_acc(xacc1, zrow, hi8, bk1f, nt1, H_ / 32,  lane);
      xacc1 = gemm_acc(xacc1, yrow, hi8, bk2f, nt1, OB_ / 32, lane);
    }

    if (t == T_ - 1) {
      int sc0 = wave * 16 + (lane & 15);
#pragma unroll
      for (int r = 0; r < 8; ++r) {
        int m = r + hi8;
        out[out_fs_base + (size_t)(m0 + m) * S_ + sc0]       = xacc0[r];
        out[out_fs_base + (size_t)(m0 + m) * S_ + sc0 + 256] = xacc1[r];
      }
    }

    __syncthreads();   // all reads of ξ (this step) complete
    {
      int sc0 = wave * 16 + (lane & 15);
#pragma unroll
      for (int r = 0; r < 8; ++r) {
        int m = r + hi8;
        xib[m * XP + sc0]       = (__bf16)xacc0[r];
        xib[m * XP + sc0 + 256] = (__bf16)xacc1[r];
      }
    }
  }
}

extern "C" void kernel_launch(void* const* d_in, const int* in_sizes, int n_in,
                              void* d_out, int out_size, void* d_ws, size_t ws_size,
                              hipStream_t stream) {
  const float* obs   = (const float*)d_in[0];
  const float* state = (const float*)d_in[1];
  const float* AK    = (const float*)d_in[2];
  const float* BK1   = (const float*)d_in[3];
  const float* BK2   = (const float*)d_in[4];
  const float* CK1   = (const float*)d_in[5];
  const float* DK1   = (const float*)d_in[6];
  const float* DK2   = (const float*)d_in[7];
  const float* CK2   = (const float*)d_in[8];
  const float* DK4   = (const float*)d_in[9];
  const float* logst = (const float*)d_in[10];
  float* out = (float*)d_out;

  // bf16 fragment-packed weights live in workspace (~3.1 MB, resident in L2)
  __bf16* ws = (__bf16*)d_ws;
  size_t off = 0;
  __bf16* ck2f = ws + off; off += (size_t)S_  * H_;
  __bf16* dk4f = ws + off; off += (size_t)OB_ * H_;
  __bf16* ck1f = ws + off; off += (size_t)S_  * AC_;
  __bf16* dk1f = ws + off; off += (size_t)H_  * AC_;
  __bf16* dk2f = ws + off; off += (size_t)OB_ * AC_;
  __bf16* akf  = ws + off; off += (size_t)S_  * S_;
  __bf16* bk1f = ws + off; off += (size_t)H_  * S_;
  __bf16* bk2f = ws + off; off += (size_t)OB_ * S_;
  (void)ws_size; (void)in_sizes; (void)n_in; (void)out_size;

  auto pack = [&](const float* W, __bf16* dst, int K, int N) {
    int total = K * N;
    pack_b_frags<<<(total + 255) / 256, 256, 0, stream>>>(W, dst, K, N);
  };
  pack(CK2, ck2f, S_, H_);
  pack(DK4, dk4f, OB_, H_);
  pack(CK1, ck1f, S_, AC_);
  pack(DK1, dk1f, H_, AC_);
  pack(DK2, dk2f, OB_, AC_);
  pack(AK,  akf,  S_, S_);
  pack(BK1, bk1f, H_, S_);
  pack(BK2, bk2f, OB_, S_);

  rnn_scan_kernel<<<B_ / 16, 512, 0, stream>>>(
      obs, state, logst, ck2f, dk4f, ck1f, dk1f, dk2f, akf, bk1f, bk2f, out);
}
